// Seq2Seq_33724083208352
// MI455X (gfx1250) — matrix operand order repaired and checked by reading, not compiled, not run on hardware
//
#include <hip/hip_runtime.h>
#include <hip/hip_bf16.h>

typedef _Float16 half_t;
typedef __attribute__((ext_vector_type(16))) _Float16 v16h;
typedef __attribute__((ext_vector_type(8)))  _Float16 v8h;
typedef __attribute__((ext_vector_type(8)))  float    v8f;

#define EDIM   512
#define GDIM   2048      // 4*E
#define BATCH  16
#define S_ENC  512
#define S_DEC  128
#define VOCAB  32000
#define KD     1024      // decoder combined K and projection K

// ---------------------------------------------------------------------------
// WMMA helpers (v_wmma_f32_16x16x32_f16), CDNA5 wave32 layouts per ISA 7.12.2
// ---------------------------------------------------------------------------
__device__ __forceinline__ v8f wmma16(v16h a, v16h b, v8f c) {
  return __builtin_amdgcn_wmma_f32_16x16x32_f16(
      /*neg_a=*/false, a, /*neg_b=*/false, b,
      /*c_mod=*/(short)0, c, /*reuse_a=*/false, /*reuse_b=*/false);
}

// A tile (16x32 f16, row-major, base points at [m0][k0], stride lda halves).
// Lane l16 = row M; hi selects K subblock {0..15} vs {16..31}.
__device__ __forceinline__ v16h ldA(const half_t* base, int lda, int l16, int hi) {
  const half_t* p = base + (long)l16 * lda + hi * 16;
  union { v16h v; v8h h[2]; } u;
  u.h[0] = *(const v8h*)(p);
  u.h[1] = *(const v8h*)(p + 8);
  return u.v;
}

// B tile (32x16 f16) from row-major weight W[N][K]; base points at [n0][k0].
// Lane l16 = column N; element j = W[n0+l16][k0 + hi*16 + j].
__device__ __forceinline__ v16h ldB(const half_t* base, int ldw, int l16, int hi) {
  return *(const v16h*)(base + (long)l16 * ldw + hi * 16);
}

// Broadcast-vector B: B[k][n] = vec[k0+k] for every column n (matvec trick).
__device__ __forceinline__ v16h ldBvec(const half_t* vec, int hi) {
  return *(const v16h*)(vec + hi * 16);
}

__device__ __forceinline__ float fast_sig(float x) {
  return 1.f / (1.f + __expf(-x));
}
__device__ __forceinline__ float fast_tanh(float x) {
  float e = __expf(-2.f * fabsf(x));
  return copysignf((1.f - e) / (1.f + e), x);
}

// ---------------------------------------------------------------------------
// Tensor Data Mover: B-tile stage (PROJ_NB rows x 64B, row stride 2048B)
// D# per ISA 8.3/8.4: group0 = {count/type/addr}, group1 = dims/strides.
// This toolchain ships the 6-arg builtin:
//   (u32x4 g0, i32x8 g1, i32x4 g2, i32x4 g3, i32x8 extra, i32 cpol)
// ---------------------------------------------------------------------------
#define PROJ_MB 64
#define PROJ_NB 128
#define PROJ_KC 32

#if __has_builtin(__builtin_amdgcn_tensor_load_to_lds)
#define USE_TDM 1
typedef unsigned int u32x4 __attribute__((ext_vector_type(4)));
typedef int i32x4 __attribute__((ext_vector_type(4)));
typedef int i32x8 __attribute__((ext_vector_type(8)));

__device__ __forceinline__ void tdm_load_b_tile(const half_t* gsrc, unsigned lds_off) {
  unsigned long long ga = (unsigned long long)(uintptr_t)gsrc;
  // group0: [1:0]=count=1 | lds_addr[63:32] | global_addr[120:64] | type=2 @127:126
  u32x4 g0 = { 1u, lds_off, (unsigned)ga,
               (unsigned)((ga >> 32) & 0x1FFFFFFull) | 0x80000000u };
  // group1: data_size=8B(3)<<16 | tensor_dim0=256(@79:48) | tensor_dim1=128(@111:80)
  //         tile_dim0=8(@127:112) | tile_dim1=128(@143:128) | dim0_stride=256(@207:160)
  i32x8 g1 = { 0x00030000, 0x01000000, 0x00800000, 0x00080000,
               0x00000080, 0x00000100, 0, 0 };
  i32x4 gz4 = { 0, 0, 0, 0 };
  i32x8 gz8 = { 0, 0, 0, 0, 0, 0, 0, 0 };
  __builtin_amdgcn_tensor_load_to_lds(g0, g1, gz4, gz4, gz8, 0);
}
#else
#define USE_TDM 0
#endif

// ---------------------------------------------------------------------------
// Prep kernels
// ---------------------------------------------------------------------------
__global__ void cvt_f16_kernel(const float* __restrict__ s, half_t* __restrict__ d, int n) {
  int i = blockIdx.x * blockDim.x + threadIdx.x;
  int stride = gridDim.x * blockDim.x;
  for (; i < n; i += stride) d[i] = (half_t)s[i];
}

__global__ void build_wd_kernel(const float* __restrict__ wih, const float* __restrict__ whh,
                                half_t* __restrict__ wd) {
  int g = blockIdx.x;  // 0..2047
  for (int k = threadIdx.x; k < KD; k += blockDim.x) {
    float v = (k < EDIM) ? wih[(long)g * EDIM + k] : whh[(long)g * EDIM + (k - EDIM)];
    wd[(long)g * KD + k] = (half_t)v;
  }
}

// X[t*16+b][e] = f16(embedding[encode_input[b][t]][e])
__global__ void gather_enc_kernel(const int* __restrict__ enc_in, const float* __restrict__ emb,
                                  half_t* __restrict__ X) {
  int row = blockIdx.x;              // row = t*16 + b
  int b = row & 15, t = row >> 4;
  int tok = enc_in[b * S_ENC + t];
  const float* src = emb + (long)tok * EDIM;
  half_t* dst = X + (long)row * EDIM;
  for (int e = threadIdx.x; e < EDIM; e += blockDim.x) dst[e] = (half_t)src[e];
}

// x0[b][e] = f16(embedding[decode_input[b][0]][e])
__global__ void gather_x0_kernel(const int* __restrict__ dec_in, const float* __restrict__ emb,
                                 half_t* __restrict__ x0) {
  int b = blockIdx.x;
  int tok = dec_in[b * S_DEC];
  for (int e = threadIdx.x; e < EDIM; e += blockDim.x)
    x0[b * EDIM + e] = (half_t)emb[(long)tok * EDIM + e];
}

// ---------------------------------------------------------------------------
// Encoder pre-gates: P[t*16+b][g] = Xenc @ w_ih_e^T + b_ih_e + b_hh_e
// M=8192, N=2048, K=512. Block = 256 thr (8 waves), each wave 4 N-tiles.
// ---------------------------------------------------------------------------
__global__ __launch_bounds__(256) void pregate_gemm_kernel(
    const half_t* __restrict__ X, const half_t* __restrict__ W,
    const float* __restrict__ bih, const float* __restrict__ bhh,
    float* __restrict__ P) {
  const int tid = threadIdx.x, wave = tid >> 5, lane = tid & 31;
  const int l16 = lane & 15, hi = (lane >> 4) & 1;
  const int m0 = blockIdx.y * 16;
  const int n0 = blockIdx.x * 512 + wave * 64;
  v8f acc[4] = {};
  for (int k0 = 0; k0 < EDIM; k0 += 32) {
    v16h a = ldA(X + (long)m0 * EDIM + k0, EDIM, l16, hi);
#pragma unroll
    for (int i = 0; i < 4; ++i) {
      v16h b = ldB(W + (long)(n0 + i * 16) * EDIM + k0, EDIM, l16, hi);
      acc[i] = wmma16(a, b, acc[i]);
    }
  }
#pragma unroll
  for (int i = 0; i < 4; ++i) {
    int n = n0 + i * 16 + l16;
    float bias = bih[n] + bhh[n];
#pragma unroll
    for (int r = 0; r < 8; ++r)
      P[(long)(m0 + r + hi * 8) * GDIM + n] = acc[i][r] + bias;
  }
}

// ---------------------------------------------------------------------------
// Encoder recurrence: single 1024-thread WG, 32 waves; wave w owns e-cols
// [16w,16w+16). c stays in VGPRs, h shared via 16KB LDS.
// ---------------------------------------------------------------------------
__global__ __launch_bounds__(1024) void encoder_kernel(
    const float* __restrict__ P, const half_t* __restrict__ Whh,
    half_t* __restrict__ enc_out,   // [s][b][e]  (s*16+b)*512+e
    half_t* __restrict__ encT,      // [b][e][s]  (b*512+e)*512+s
    half_t* __restrict__ hT, float* __restrict__ cT) {
  __shared__ __align__(64) half_t hbuf[BATCH * EDIM];
  const int tid = threadIdx.x, wave = tid >> 5, lane = tid & 31;
  const int l16 = lane & 15, hi = (lane >> 4) & 1;
  const int e0 = wave * 16;

  for (int i = tid; i < BATCH * EDIM; i += 1024) hbuf[i] = (half_t)0.f;
  v8f c = {};
  __syncthreads();

  for (int t = 0; t < S_ENC; ++t) {
    v8f acc[4];
#pragma unroll
    for (int q = 0; q < 4; ++q)
#pragma unroll
      for (int r = 0; r < 8; ++r)
        acc[q][r] = P[(long)(t * BATCH + r + hi * 8) * GDIM + q * EDIM + e0 + l16];

    for (int k0 = 0; k0 < EDIM; k0 += 32) {
      v16h a = ldA(hbuf + k0, EDIM, l16, hi);
#pragma unroll
      for (int q = 0; q < 4; ++q) {
        v16h b = ldB(Whh + (long)(q * EDIM + e0) * EDIM + k0, EDIM, l16, hi);
        acc[q] = wmma16(a, b, acc[q]);
      }
    }
    __syncthreads();  // all reads of hbuf finished before overwrite
#pragma unroll
    for (int r = 0; r < 8; ++r) {
      float ig = fast_sig(acc[0][r]);
      float fg = fast_sig(acc[1][r]);
      float gg = fast_tanh(acc[2][r]);
      float og = fast_sig(acc[3][r]);
      float cn = fg * c[r] + ig * gg;
      c[r] = cn;
      half_t hh = (half_t)(og * fast_tanh(cn));
      int m = r + hi * 8, e = e0 + l16;
      hbuf[m * EDIM + e] = hh;
      enc_out[(long)(t * BATCH + m) * EDIM + e] = hh;
      encT[((long)m * EDIM + e) * S_ENC + t] = hh;
    }
    __syncthreads();  // writes visible before next step's A loads
  }
#pragma unroll
  for (int r = 0; r < 8; ++r) {
    int m = r + hi * 8, e = e0 + l16;
    hT[m * EDIM + e] = hbuf[m * EDIM + e];
    cT[m * EDIM + e] = c[r];
  }
}

// ---------------------------------------------------------------------------
// Decoder: LSTM (combined [x;h] K=1024 GEMM) + WMMA attention + writes
// dec_out[b*128+t][0:512]=h, [512:1024]=ctx (f16), single 1024-thread WG.
// ---------------------------------------------------------------------------
__global__ __launch_bounds__(1024) void decoder_kernel(
    const half_t* __restrict__ Wd, const float* __restrict__ bih, const float* __restrict__ bhh,
    const half_t* __restrict__ x0, const half_t* __restrict__ hT, const float* __restrict__ cT,
    const half_t* __restrict__ enc_out, const half_t* __restrict__ encT,
    half_t* __restrict__ dec_out) {
  __shared__ __align__(64) half_t xh[BATCH * KD];        // 32KB: [b][0:512]=x, [512:1024]=h
  __shared__ __align__(64) float scores[BATCH * S_ENC];  // 32KB, low 16KB aliased as f16 attn
  half_t* attn = (half_t*)scores;

  const int tid = threadIdx.x, wave = tid >> 5, lane = tid & 31;
  const int l16 = lane & 15, hi = (lane >> 4) & 1;
  const int e0 = wave * 16;

  for (int i = tid; i < BATCH * EDIM; i += 1024) {
    int b = i >> 9, e = i & 511;
    xh[b * KD + e] = x0[i];
    xh[b * KD + EDIM + e] = hT[i];
  }
  v8f c;
  float bias_q[4];
  {
    int e = e0 + l16;
#pragma unroll
    for (int r = 0; r < 8; ++r) c[r] = cT[(r + hi * 8) * EDIM + e];
#pragma unroll
    for (int q = 0; q < 4; ++q) bias_q[q] = bih[q * EDIM + e] + bhh[q * EDIM + e];
  }
  __syncthreads();

  for (int t = 0; t < S_DEC; ++t) {
    // ---- gates = [x;h] @ Wd^T + bias ----
    v8f acc[4];
#pragma unroll
    for (int q = 0; q < 4; ++q)
#pragma unroll
      for (int r = 0; r < 8; ++r) acc[q][r] = bias_q[q];
    for (int k0 = 0; k0 < KD; k0 += 32) {
      v16h a = ldA(xh + k0, KD, l16, hi);
#pragma unroll
      for (int q = 0; q < 4; ++q) {
        v16h b = ldB(Wd + (long)(q * EDIM + e0) * KD + k0, KD, l16, hi);
        acc[q] = wmma16(a, b, acc[q]);
      }
    }
    __syncthreads();
    // ---- pointwise LSTM; h feeds both x and h slots for next step ----
#pragma unroll
    for (int r = 0; r < 8; ++r) {
      float ig = fast_sig(acc[0][r]);
      float fg = fast_sig(acc[1][r]);
      float gg = fast_tanh(acc[2][r]);
      float og = fast_sig(acc[3][r]);
      float cn = fg * c[r] + ig * gg;
      c[r] = cn;
      half_t hh = (half_t)(og * fast_tanh(cn));
      int m = r + hi * 8, e = e0 + l16;
      xh[m * KD + e] = hh;
      xh[m * KD + EDIM + e] = hh;
      dec_out[(long)(m * S_DEC + t) * KD + e] = hh;
    }
    __syncthreads();
    // ---- scores[b][s] = h[b] . enc_out[s][b]  (WMMA, broadcast-B) ----
    for (int j = wave; j < BATCH * 32; j += 32) {
      int b = j >> 5, s0 = (j & 31) << 4;
      v8f sc = {};
      for (int k0 = 0; k0 < EDIM; k0 += 32) {
        v16h a = ldA(enc_out + (long)(s0 * BATCH + b) * EDIM + k0, BATCH * EDIM, l16, hi);
        v16h bb = ldBvec(xh + b * KD + EDIM + k0, hi);
        sc = wmma16(a, bb, sc);
      }
      if (l16 == 0) {
#pragma unroll
        for (int r = 0; r < 8; ++r) scores[b * S_ENC + s0 + r + hi * 8] = sc[r];
      }
    }
    __syncthreads();
    // ---- softmax (waves 0..15, one row each); read phase then write phase
    //      so the f16 attn alias never races the f32 score reads ----
    float v[16];
    float inv = 0.f;
    if (wave < 16) {
      int b = wave;
      float mx = -1e30f;
#pragma unroll
      for (int i = 0; i < 16; ++i) {
        v[i] = scores[b * S_ENC + lane + 32 * i];
        mx = fmaxf(mx, v[i]);
      }
      for (int off = 16; off > 0; off >>= 1) mx = fmaxf(mx, __shfl_xor(mx, off, 32));
      float s = 0.f;
#pragma unroll
      for (int i = 0; i < 16; ++i) { v[i] = __expf(v[i] - mx); s += v[i]; }
      for (int off = 16; off > 0; off >>= 1) s += __shfl_xor(s, off, 32);
      inv = 1.f / s;
    }
    __syncthreads();
    if (wave < 16) {
      int b = wave;
#pragma unroll
      for (int i = 0; i < 16; ++i) attn[b * S_ENC + lane + 32 * i] = (half_t)(v[i] * inv);
    }
    __syncthreads();
    // ---- ctx[b][e] = sum_s attn[b][s] * encT[b][e][s]  (WMMA, broadcast-B) ----
    for (int j = wave; j < BATCH * 32; j += 32) {
      int b = j >> 5, ee = (j & 31) << 4;
      v8f cx = {};
      for (int k0 = 0; k0 < S_ENC; k0 += 32) {
        v16h a = ldA(encT + ((long)b * EDIM + ee) * S_ENC + k0, S_ENC, l16, hi);
        v16h bb = ldBvec(attn + b * S_ENC + k0, hi);
        cx = wmma16(a, bb, cx);
      }
      if (l16 == 0) {
#pragma unroll
        for (int r = 0; r < 8; ++r)
          dec_out[(long)(b * S_DEC + t) * KD + EDIM + ee + r + hi * 8] = (half_t)cx[r];
      }
    }
    __syncthreads();
  }
}

// ---------------------------------------------------------------------------
// Projection: logits = dec_out(2048x1024) @ w_proj^T(32000x1024) + b_proj
// Block = 256 thr (8 waves) computing 64M x 128N; B tile staged in LDS
// (TDM double-buffered when available), shared by all 8 waves -> 8x less
// L2 traffic on the 65MB weight matrix. Waves: 4(M) x 2(N), 4 acc tiles.
// ---------------------------------------------------------------------------
__global__ __launch_bounds__(256) void proj_gemm_kernel(
    const half_t* __restrict__ Adec, const half_t* __restrict__ W,
    const float* __restrict__ bias, float* __restrict__ out) {
  __shared__ __align__(64) half_t Bsh[2][PROJ_NB * PROJ_KC];  // 2 x 8KB
  const int tid = threadIdx.x, wave = tid >> 5, lane = tid & 31;
  const int l16 = lane & 15, hi = (lane >> 4) & 1;
  const int wm = wave & 3, wn = wave >> 2;
  const int m0 = blockIdx.y * PROJ_MB + wm * 16;
  const int nblk = blockIdx.x * PROJ_NB;
  const int nstages = KD / PROJ_KC;  // 32

  auto load_stage = [&](int s, int buf) {
#if USE_TDM
    if (wave == 0)
      tdm_load_b_tile(W + (long)nblk * KD + s * PROJ_KC,
                      (unsigned)(uintptr_t)(void*)&Bsh[buf][0]);
#else
    for (int i = tid; i < PROJ_NB * PROJ_KC / 8; i += 256) {
      int row = (i * 8) / PROJ_KC, col = (i * 8) % PROJ_KC;
      *(v8h*)&Bsh[buf][row * PROJ_KC + col] =
          *(const v8h*)(W + (long)(nblk + row) * KD + s * PROJ_KC + col);
    }
#endif
  };

  v8f acc[4] = {};
  load_stage(0, 0);
  for (int s = 0; s < nstages; ++s) {
    const int cur = s & 1;
    if (s + 1 < nstages) load_stage(s + 1, cur ^ 1);
#if USE_TDM
    if (wave == 0) {
      if (s + 1 < nstages) __builtin_amdgcn_s_wait_tensorcnt(1);  // stage s landed
      else                 __builtin_amdgcn_s_wait_tensorcnt(0);
    }
#endif
    if (s + 1 < nstages)
      __builtin_prefetch(Adec + (long)m0 * KD + (s + 1) * PROJ_KC, 0, 0);
    __syncthreads();
    v16h a = ldA(Adec + (long)m0 * KD + s * PROJ_KC, KD, l16, hi);
#pragma unroll
    for (int i = 0; i < 4; ++i) {
      v16h b = ldB(&Bsh[cur][(wn * 64 + i * 16) * PROJ_KC], PROJ_KC, l16, hi);
      acc[i] = wmma16(a, b, acc[i]);
    }
    __syncthreads();
  }
#pragma unroll
  for (int i = 0; i < 4; ++i) {
    int n = nblk + wn * 64 + i * 16 + l16;
    float bv = bias[n];
#pragma unroll
    for (int r = 0; r < 8; ++r)
      out[(long)(m0 + r + hi * 8) * VOCAB + n] = acc[i][r] + bv;
  }
}

// ---------------------------------------------------------------------------
extern "C" void kernel_launch(void* const* d_in, const int* in_sizes, int n_in,
                              void* d_out, int out_size, void* d_ws, size_t ws_size,
                              hipStream_t stream) {
  (void)in_sizes; (void)n_in; (void)out_size; (void)ws_size;
  const int*   enc_in = (const int*)d_in[0];
  const int*   dec_in = (const int*)d_in[1];
  const float* emb    = (const float*)d_in[2];
  const float* w_ih_e = (const float*)d_in[3];
  const float* w_hh_e = (const float*)d_in[4];
  const float* b_ih_e = (const float*)d_in[5];
  const float* b_hh_e = (const float*)d_in[6];
  const float* w_ih_d = (const float*)d_in[7];
  const float* w_hh_d = (const float*)d_in[8];
  const float* b_ih_d = (const float*)d_in[9];
  const float* b_hh_d = (const float*)d_in[10];
  const float* w_proj = (const float*)d_in[11];
  const float* b_proj = (const float*)d_in[12];
  float* logits = (float*)d_out;

  // workspace bump allocator (~170 MB total)
  char* p = (char*)d_ws;
  auto alloc = [&](size_t bytes) -> char* {
    char* r = p;
    p += (bytes + 255) & ~(size_t)255;
    return r;
  };
  half_t* Xenc   = (half_t*)alloc((size_t)S_ENC * BATCH * EDIM * 2);   // 8 MB
  half_t* WiheF  = (half_t*)alloc((size_t)GDIM * EDIM * 2);            // 2 MB
  half_t* WhheF  = (half_t*)alloc((size_t)GDIM * EDIM * 2);            // 2 MB
  half_t* WdF    = (half_t*)alloc((size_t)GDIM * KD * 2);              // 4 MB
  half_t* WprojF = (half_t*)alloc((size_t)VOCAB * KD * 2);             // 65.5 MB
  float*  P      = (float*) alloc((size_t)S_ENC * BATCH * GDIM * 4);   // 64 MB
  half_t* encOut = (half_t*)alloc((size_t)S_ENC * BATCH * EDIM * 2);   // 8 MB
  half_t* encT   = (half_t*)alloc((size_t)BATCH * EDIM * S_ENC * 2);   // 8 MB
  half_t* hTbuf  = (half_t*)alloc((size_t)BATCH * EDIM * 2);
  float*  cTbuf  = (float*) alloc((size_t)BATCH * EDIM * 4);
  half_t* x0buf  = (half_t*)alloc((size_t)BATCH * EDIM * 2);
  half_t* decOut = (half_t*)alloc((size_t)BATCH * S_DEC * KD * 2);     // 4 MB

  // prep / precision conversion
  cvt_f16_kernel<<<1024, 256, 0, stream>>>(w_ih_e, WiheF, GDIM * EDIM);
  cvt_f16_kernel<<<1024, 256, 0, stream>>>(w_hh_e, WhheF, GDIM * EDIM);
  cvt_f16_kernel<<<4096, 256, 0, stream>>>(w_proj, WprojF, VOCAB * KD);
  build_wd_kernel<<<GDIM, 256, 0, stream>>>(w_ih_d, w_hh_d, WdF);
  gather_enc_kernel<<<S_ENC * BATCH, 256, 0, stream>>>(enc_in, emb, Xenc);
  gather_x0_kernel<<<BATCH, 256, 0, stream>>>(dec_in, emb, x0buf);

  // hoisted encoder input GEMM (full-chip parallel)
  pregate_gemm_kernel<<<dim3(GDIM / 512, (S_ENC * BATCH) / 16), 256, 0, stream>>>(
      Xenc, WiheF, b_ih_e, b_hh_e, P);

  // sequential recurrences (single WGP each, 32 waves)
  encoder_kernel<<<1, 1024, 0, stream>>>(P, WhheF, encOut, encT, hTbuf, cTbuf);
  decoder_kernel<<<1, 1024, 0, stream>>>(WdF, b_ih_d, b_hh_d, x0buf, hTbuf, cTbuf,
                                         encOut, encT, decOut);

  // dominant projection GEMM (134 GFLOP, full-chip parallel, LDS/TDM B reuse)
  proj_gemm_kernel<<<dim3(VOCAB / PROJ_NB, (BATCH * S_DEC) / PROJ_MB), 256, 0, stream>>>(
      decOut, WprojF, b_proj, logits);
}
